// SparseResBlockDownsample3d_858993459497
// MI455X (gfx1250) — matrix operand order repaired and checked by reading, not compiled, not run on hardware
//
#include <hip/hip_runtime.h>
#include <hip/hip_bf16.h>

typedef unsigned short u16;
typedef __bf16 bf16_t;
typedef bf16_t v16bf __attribute__((ext_vector_type(16)));
typedef u16    v16u  __attribute__((ext_vector_type(16)));
typedef u16    v8u   __attribute__((ext_vector_type(8)));
typedef float  v8f   __attribute__((ext_vector_type(8)));

#define C_IN  64
#define C_OUT 128
#define CT    64          // cells (output rows) per block
#define PITCH1 72         // padded LDS row pitch (u16) for conv1 tiles (144 B)
#define PITCH2 136        // padded LDS row pitch (u16) for conv2 tiles (272 B)
#define LN_EPS 1e-6f

// butterfly-add via ds_swizzle (group-of-32 xor pattern, and_mask=0x1F)
#define SWZ_ADD(x, imm)                                                        \
  ((x) + __builtin_bit_cast(float, __builtin_amdgcn_ds_swizzle(                \
             __builtin_bit_cast(int, (x)), (imm))))

// ---------- helpers ----------

__device__ __forceinline__ u16 f2bf(float f) {
  unsigned u = __builtin_bit_cast(unsigned, f);
  u += 0x7FFFu + ((u >> 16) & 1u);        // round-to-nearest-even
  return (u16)(u >> 16);
}

__device__ __forceinline__ v8f wmma_bf16(v16bf a, v16bf b, v8f c) {
  return __builtin_amdgcn_wmma_f32_16x16x32_bf16(false, a, false, b, (short)0, c,
                                                 false, false);
}

// A-fragment (16x32 bf16), contiguous k order:
//  lanes 0-15 : row = lane,    elems = K{kb..kb+15}, kb = 32*ks + 0
//  lanes16-31 : row = lane-16, same but kb = 32*ks + 16
__device__ __forceinline__ v16bf load_a_frag(const u16* rowbase, int kb) {
  v8u c0 = *(const v8u*)(rowbase + kb);
  v8u c1 = *(const v8u*)(rowbase + kb + 8);
  v16u r;
#pragma unroll
  for (int i = 0; i < 8; ++i) { r[i] = c0[i]; r[8 + i] = c1[i]; }
  return __builtin_bit_cast(v16bf, r);
}

// B-fragment: pre-swizzled so each lane reads 16 contiguous bf16 (32B)
__device__ __forceinline__ v16bf load_b_frag(const u16* fragbase, int lane) {
  v16u r = *(const v16u*)(fragbase + lane * 16);
  return __builtin_bit_cast(v16bf, r);
}

// CDNA5 async DMA, GVS form: 64-bit uniform base in SGPRs + 32-bit VGPR byte
// offset; 16 bytes per lane direct to LDS, tracked by ASYNCcnt.
// Low 32 bits of a generic __shared__ pointer are the LDS byte offset.
__device__ __forceinline__ void async_copy_b128(void* lds_dst, const void* sbase,
                                                unsigned goff) {
  unsigned loff = (unsigned)(unsigned long long)lds_dst;
  asm volatile("global_load_async_to_lds_b128 %0, %1, %2"
               :: "v"(loff), "v"(goff), "s"(sbase) : "memory");
}

__device__ __forceinline__ void wait_async0() {
  asm volatile("s_wait_asynccnt 0x0" ::: "memory");
}

// ---------- kernel 1: weight bf16 conversion + B-fragment swizzle,
//                      plus zeroing of the sentinel rows of hd/h1 ----------
__global__ void k_prep_weights(const float* __restrict__ w1,
                               const float* __restrict__ w2,
                               const float* __restrict__ wsk,
                               u16* __restrict__ w1s,
                               u16* __restrict__ w2s,
                               u16* __restrict__ wss,
                               u16* __restrict__ hd,
                               u16* __restrict__ h1, int M) {
  int id = blockIdx.x * blockDim.x + threadIdx.x;
  const int n1 = 27 * C_IN * C_OUT;
  const int n2 = 27 * C_OUT * C_OUT;
  const int n3 = C_IN * C_OUT;
  if (id < n1) {
    int t = id / (C_IN * C_OUT);
    int rem = id - t * (C_IN * C_OUT);
    int kin = rem / C_OUT, n = rem % C_OUT;
    int ks = kin >> 5, kk = kin & 31;
    int lane = (n & 15) + ((kk >> 4) << 4);
    int frag = (t * 2 + ks) * 8 + (n >> 4);
    w1s[frag * 512 + lane * 16 + (kk & 15)] = f2bf(w1[id]);
  } else if (id < n1 + n2) {
    int j = id - n1;
    int t = j / (C_OUT * C_OUT);
    int rem = j - t * (C_OUT * C_OUT);
    int kin = rem / C_OUT, n = rem % C_OUT;
    int ks = kin >> 5, kk = kin & 31;
    int lane = (n & 15) + ((kk >> 4) << 4);
    int frag = (t * 4 + ks) * 8 + (n >> 4);
    w2s[frag * 512 + lane * 16 + (kk & 15)] = f2bf(w2[j]);
  } else if (id < n1 + n2 + n3) {
    int j = id - n1 - n2;
    int kin = j / C_OUT, n = j % C_OUT;
    int ks = kin >> 5, kk = kin & 31;
    int lane = (n & 15) + ((kk >> 4) << 4);
    int frag = ks * 8 + (n >> 4);
    wss[frag * 512 + lane * 16 + (kk & 15)] = f2bf(wsk[j]);
  } else if (id < n1 + n2 + n3 + C_IN) {
    hd[(size_t)M * C_IN + (id - n1 - n2 - n3)] = 0;         // sentinel row
  } else if (id < n1 + n2 + n3 + C_IN + C_OUT) {
    h1[(size_t)M * C_OUT + (id - n1 - n2 - n3 - C_IN)] = 0; // sentinel row
  }
}

// ---------- kernel 2: LN(affine)+SiLU per child row, mean-pool 2->1 ----------
__global__ __launch_bounds__(256) void k_pool(const float* __restrict__ feats,
                                              const float* __restrict__ lnw,
                                              const float* __restrict__ lnb,
                                              const int* __restrict__ counts,
                                              u16* __restrict__ hd,
                                              u16* __restrict__ xd, int M) {
  int wv = threadIdx.x >> 5, lane = threadIdx.x & 31;
  int m = blockIdx.x * 8 + wv;
  if (m >= M) return;
  float inv = 1.0f / (float)counts[m];
  float w0 = lnw[lane], w1 = lnw[lane + 32];
  float b0 = lnb[lane], b1 = lnb[lane + 32];
  float hs0 = 0.f, hs1 = 0.f, xs0 = 0.f, xs1 = 0.f;
#pragma unroll
  for (int c = 0; c < 2; ++c) {
    const float* row = feats + (size_t)(2 * m + c) * C_IN;
    float x0 = row[lane], x1 = row[lane + 32];
    float s = x0 + x1;
    s = SWZ_ADD(s, 0x041F); s = SWZ_ADD(s, 0x081F); s = SWZ_ADD(s, 0x101F);
    s = SWZ_ADD(s, 0x201F); s = SWZ_ADD(s, 0x401F);
    float mu = s * (1.0f / 64.0f);
    float d0 = x0 - mu, d1 = x1 - mu;
    float q = d0 * d0 + d1 * d1;
    q = SWZ_ADD(q, 0x041F); q = SWZ_ADD(q, 0x081F); q = SWZ_ADD(q, 0x101F);
    q = SWZ_ADD(q, 0x201F); q = SWZ_ADD(q, 0x401F);
    float rstd = rsqrtf(q * (1.0f / 64.0f) + LN_EPS);
    float y0 = d0 * rstd * w0 + b0;
    float y1 = d1 * rstd * w1 + b1;
    y0 = y0 / (1.0f + __expf(-y0));
    y1 = y1 / (1.0f + __expf(-y1));
    hs0 += y0; hs1 += y1; xs0 += x0; xs1 += x1;
  }
  size_t base = (size_t)m * C_IN;
  hd[base + lane]      = f2bf(hs0 * inv);
  hd[base + lane + 32] = f2bf(hs1 * inv);
  xd[base + lane]      = f2bf(xs0 * inv);
  xd[base + lane + 32] = f2bf(xs1 * inv);
}

// ---------- kernel 3: conv1 (27 taps, K=64) + LN(no affine) + SiLU ----------
__global__ __launch_bounds__(256) void k_conv1(const u16* __restrict__ hd,
                                               const int* __restrict__ nbr,
                                               const u16* __restrict__ w1s,
                                               const float* __restrict__ b1,
                                               u16* __restrict__ h1, int M) {
  __shared__ alignas(16) u16 tileA[2][CT * PITCH1];  // 2 x 9 KB (padded)
  __shared__ int idxs[CT * 27];                      // 6.75 KB
  __shared__ float part_s[8][CT];
  __shared__ float part_q[8][CT];
  __shared__ float tot_s[CT], tot_q[CT];

  int tid = threadIdx.x;
  int wv = tid >> 5, lane = tid & 31;
  int half = lane >> 4;
  int kb_half = half ? 16 : 0;
  int colq = lane & 15;
  int ntile = wv;
  int cell0 = blockIdx.x * CT;

  // fixed per-thread gather geometry: 2 chunks of 16B, 8 chunks per 128B row
  int r0 = tid >> 3, c0 = tid & 7;          // chunk tid
  int r1 = r0 + 32;                          // chunk tid+256 (same c)

  for (int j = tid; j < CT * 27; j += 256) idxs[j] = nbr[(size_t)cell0 * 27 + j];
  __syncthreads();

  auto gather = [&](int k, u16* dst) {
    unsigned g0 = ((unsigned)idxs[r0 * 27 + k] << 7) + (unsigned)(c0 * 16);
    unsigned g1 = ((unsigned)idxs[r1 * 27 + k] << 7) + (unsigned)(c0 * 16);
    async_copy_b128(dst + (size_t)r0 * PITCH1 + c0 * 8, hd, g0);
    async_copy_b128(dst + (size_t)r1 * PITCH1 + c0 * 8, hd, g1);
  };

  v8f acc[4] = {};

  gather(0, tileA[0]);
  wait_async0();
  __syncthreads();

  for (int k = 0; k < 27; ++k) {
    const u16* cur = tileA[k & 1];
    if (k < 26) gather(k + 1, tileA[(k + 1) & 1]);   // overlap DMA with WMMA
#pragma unroll
    for (int ks = 0; ks < 2; ++ks) {
      v16bf b = load_b_frag(w1s + (size_t)((k * 2 + ks) * 8 + ntile) * 512, lane);
      int kb = ks * 32 + kb_half;
#pragma unroll
      for (int t = 0; t < 4; ++t) {
        v16bf a = load_a_frag(cur + (t * 16 + colq) * PITCH1, kb);
        acc[t] = wmma_bf16(a, b, acc[t]);
      }
    }
    wait_async0();
    __syncthreads();
  }

  // bias (each lane's 8 acc elements share one output column)
  float bias = b1[ntile * 16 + colq];
#pragma unroll
  for (int t = 0; t < 4; ++t)
#pragma unroll
    for (int v = 0; v < 8; ++v) acc[t][v] += bias;

  // per-row partial sums over this wave's 16 columns (xor butterfly in halves)
#pragma unroll
  for (int t = 0; t < 4; ++t) {
#pragma unroll
    for (int v = 0; v < 8; ++v) {
      float s = acc[t][v];
      float q = s * s;
      s = SWZ_ADD(s, 0x041F); s = SWZ_ADD(s, 0x081F);
      s = SWZ_ADD(s, 0x101F); s = SWZ_ADD(s, 0x201F);
      q = SWZ_ADD(q, 0x041F); q = SWZ_ADD(q, 0x081F);
      q = SWZ_ADD(q, 0x101F); q = SWZ_ADD(q, 0x201F);
      if (colq == 0) {
        int row = t * 16 + v + 8 * half;
        part_s[wv][row] = s;
        part_q[wv][row] = q;
      }
    }
  }
  __syncthreads();
  if (tid < CT) {
    float s = 0.f, q = 0.f;
#pragma unroll
    for (int w = 0; w < 8; ++w) { s += part_s[w][tid]; q += part_q[w][tid]; }
    tot_s[tid] = s; tot_q[tid] = q;
  }
  __syncthreads();

#pragma unroll
  for (int t = 0; t < 4; ++t) {
#pragma unroll
    for (int v = 0; v < 8; ++v) {
      int row = t * 16 + v + 8 * half;
      float mu = tot_s[row] * (1.0f / 128.0f);
      float var = tot_q[row] * (1.0f / 128.0f) - mu * mu;
      float y = (acc[t][v] - mu) * rsqrtf(var + LN_EPS);
      y = y / (1.0f + __expf(-y));
      int cell = cell0 + row;
      if (cell < M)
        h1[(size_t)cell * C_OUT + ntile * 16 + colq] = f2bf(y);
    }
  }
}

// ---------- kernel 4: conv2 (27 taps, K=128) + skip GEMM + biases ----------
__global__ __launch_bounds__(256) void k_conv2(const u16* __restrict__ h1,
                                               const u16* __restrict__ xd,
                                               const int* __restrict__ nbr,
                                               const u16* __restrict__ w2s,
                                               const u16* __restrict__ wss,
                                               const float* __restrict__ b2,
                                               const float* __restrict__ bsk,
                                               float* __restrict__ out, int M) {
  __shared__ alignas(16) u16 tileA[2][CT * PITCH2];  // 2 x 17 KB (padded)
  __shared__ int idxs[CT * 27];                      // 6.75 KB

  int tid = threadIdx.x;
  int wv = tid >> 5, lane = tid & 31;
  int half = lane >> 4;
  int kb_half = half ? 16 : 0;
  int colq = lane & 15;
  int ntile = wv;
  int cell0 = blockIdx.x * CT;

  // 4 chunks of 16B per thread, 16 chunks per 256B row
  int r0 = tid >> 4, c0 = tid & 15;   // rows r0, r0+16, r0+32, r0+48

  for (int j = tid; j < CT * 27; j += 256) idxs[j] = nbr[(size_t)cell0 * 27 + j];
  __syncthreads();

  auto gather = [&](int k, u16* dst) {
#pragma unroll
    for (int i = 0; i < 4; ++i) {
      int r = r0 + 16 * i;
      unsigned g = ((unsigned)idxs[r * 27 + k] << 8) + (unsigned)(c0 * 16);
      async_copy_b128(dst + (size_t)r * PITCH2 + c0 * 8, h1, g);
    }
  };

  v8f acc[4] = {};

  gather(0, tileA[0]);
  wait_async0();
  __syncthreads();

  for (int k = 0; k < 27; ++k) {
    const u16* cur = tileA[k & 1];
    if (k < 26) gather(k + 1, tileA[(k + 1) & 1]);   // overlap DMA with WMMA
#pragma unroll
    for (int ks = 0; ks < 4; ++ks) {
      v16bf b = load_b_frag(w2s + (size_t)((k * 4 + ks) * 8 + ntile) * 512, lane);
      int kb = ks * 32 + kb_half;
#pragma unroll
      for (int t = 0; t < 4; ++t) {
        v16bf a = load_a_frag(cur + (t * 16 + colq) * PITCH2, kb);
        acc[t] = wmma_bf16(a, b, acc[t]);
      }
    }
    wait_async0();
    __syncthreads();
  }

  // skip path: two extra taps, A rows are contiguous xd rows (no gather)
#pragma unroll
  for (int ks = 0; ks < 2; ++ks) {
    v16bf b = load_b_frag(wss + (size_t)(ks * 8 + ntile) * 512, lane);
    int kb = ks * 32 + kb_half;
#pragma unroll
    for (int t = 0; t < 4; ++t) {
      int row = cell0 + t * 16 + colq;
      if (row >= M) row = M - 1;          // safe clamp (M % CT == 0 in practice)
      v16bf a = load_a_frag(xd + (size_t)row * C_IN, kb);
      acc[t] = wmma_bf16(a, b, acc[t]);
    }
  }

  float bias = b2[ntile * 16 + colq] + bsk[ntile * 16 + colq];
#pragma unroll
  for (int t = 0; t < 4; ++t) {
#pragma unroll
    for (int v = 0; v < 8; ++v) {
      int row = t * 16 + v + 8 * half;
      int cell = cell0 + row;
      if (cell < M)
        out[(size_t)cell * C_OUT + ntile * 16 + colq] = acc[t][v] + bias;
    }
  }
}

// ---------- launch ----------
extern "C" void kernel_launch(void* const* d_in, const int* in_sizes, int n_in,
                              void* d_out, int out_size, void* d_ws, size_t ws_size,
                              hipStream_t stream) {
  const float* feats   = (const float*)d_in[0];
  const float* ln1_w   = (const float*)d_in[1];
  const float* ln1_b   = (const float*)d_in[2];
  const float* conv1_W = (const float*)d_in[3];
  const float* conv1_b = (const float*)d_in[4];
  const float* conv2_W = (const float*)d_in[5];
  const float* conv2_b = (const float*)d_in[6];
  const float* skip_W  = (const float*)d_in[7];
  const float* skip_b  = (const float*)d_in[8];
  const int*   counts  = (const int*)d_in[10];
  const int*   nbr_idx = (const int*)d_in[11];
  float* out = (float*)d_out;

  int M = in_sizes[10];                    // counts has M elements

  char* ws = (char*)d_ws;
  size_t off = 0;
  auto alloc = [&](size_t bytes) -> void* {
    void* p = ws + off;
    off = (off + bytes + 255) & ~(size_t)255;
    return p;
  };
  u16* hd  = (u16*)alloc((size_t)(M + 1) * C_IN * sizeof(u16));   // +sentinel row
  u16* xd  = (u16*)alloc((size_t)M * C_IN * sizeof(u16));
  u16* h1  = (u16*)alloc((size_t)(M + 1) * C_OUT * sizeof(u16));  // +sentinel row
  u16* w1s = (u16*)alloc((size_t)27 * C_IN * C_OUT * sizeof(u16));
  u16* w2s = (u16*)alloc((size_t)27 * C_OUT * C_OUT * sizeof(u16));
  u16* wss = (u16*)alloc((size_t)C_IN * C_OUT * sizeof(u16));

  int nprep = 27 * C_IN * C_OUT + 27 * C_OUT * C_OUT + C_IN * C_OUT + C_IN + C_OUT;
  k_prep_weights<<<(nprep + 255) / 256, 256, 0, stream>>>(conv1_W, conv2_W, skip_W,
                                                          w1s, w2s, wss, hd, h1, M);
  k_pool<<<(M + 7) / 8, 256, 0, stream>>>(feats, ln1_w, ln1_b, counts, hd, xd, M);
  int nblk = (M + CT - 1) / CT;
  k_conv1<<<nblk, 256, 0, stream>>>(hd, nbr_idx, w1s, conv1_b, h1, M);
  k_conv2<<<nblk, 256, 0, stream>>>(h1, xd, nbr_idx, w2s, wss, conv2_b, skip_b, out, M);
}
